// ContrastiveLoss_64613488001156
// MI455X (gfx1250) — compile-verified, matrix operand order checked by reading, not run on hardware
//
#include <hip/hip_runtime.h>

// ---------------------------------------------------------------------------
// ContrastiveLoss (soft-DTW VSE loss) for MI455X / gfx1250
//   B=32, M=128, N=128, C=1024, gamma=0.1, margin=0, BIG=1e9
// Pipeline:
//   K1 norm_split_bf16_kernel : L2-normalize rows of im/s, emit SPLIT bf16
//                               (hi + residual lo) operands in workspace
//   K2 pair_dtw_kernel        : per (i,j) pair: 128x128 error-compensated
//                               bf16 WMMA GEMM (hi*hi + hi*lo + lo*hi) ->
//                               LDS cost tile, then anti-diagonal soft-DTW
//                               wavefront in LDS; scores[p] = -R[la][lb]/100
//   K3 loss_kernel            : contrastive hinge reduction -> scalar out
// ---------------------------------------------------------------------------

typedef __attribute__((ext_vector_type(8)))  float  v8f;
typedef __attribute__((ext_vector_type(8)))  __bf16 v8bf;
typedef __attribute__((ext_vector_type(16))) __bf16 v16bf;

#define C_DIM 1024
#define BDIM  32
#define MDIM  128
#define NDIM  128
#define BIGF  1000000000.0f
#define DSTRIDE 130   // 128 cols padded; diagonal read stride 129 == 1 mod 64 banks

// ---------------------------------------------------------------------------
// K1: per-row L2 normalize + f32 -> split bf16 (hi + lo residual).
// grid = 8192 rows (4096 im rows then 4096 s rows), block = 256.
// ---------------------------------------------------------------------------
__global__ __launch_bounds__(256) void norm_split_bf16_kernel(
    const float* __restrict__ im, const float* __restrict__ s,
    __bf16* __restrict__ imb_hi, __bf16* __restrict__ imb_lo,
    __bf16* __restrict__ sb_hi,  __bf16* __restrict__ sb_lo) {
  __shared__ float red[256];
  const int row = blockIdx.x;
  const float* src;
  __bf16* dhi;
  __bf16* dlo;
  if (row < BDIM * MDIM) {
    src = im + (size_t)row * C_DIM;
    dhi = imb_hi + (size_t)row * C_DIM;
    dlo = imb_lo + (size_t)row * C_DIM;
  } else {
    const int r2 = row - BDIM * MDIM;
    src = s + (size_t)r2 * C_DIM;
    dhi = sb_hi + (size_t)r2 * C_DIM;
    dlo = sb_lo + (size_t)r2 * C_DIM;
  }
  const int t = threadIdx.x;
  const float4 v = ((const float4*)src)[t];          // 256 * 4 = 1024 elements
  red[t] = v.x * v.x + v.y * v.y + v.z * v.z + v.w * v.w;
  __syncthreads();
#pragma unroll
  for (int k = 128; k > 0; k >>= 1) {
    if (t < k) red[t] += red[t + k];
    __syncthreads();
  }
  const float sc = 1.0f / fmaxf(sqrtf(red[0]), 1e-12f);
  const float f0 = v.x * sc, f1 = v.y * sc, f2 = v.z * sc, f3 = v.w * sc;
  const __bf16 h0 = (__bf16)f0, h1 = (__bf16)f1, h2 = (__bf16)f2, h3 = (__bf16)f3;
  dhi[4 * t + 0] = h0;
  dhi[4 * t + 1] = h1;
  dhi[4 * t + 2] = h2;
  dhi[4 * t + 3] = h3;
  dlo[4 * t + 0] = (__bf16)(f0 - (float)h0);
  dlo[4 * t + 1] = (__bf16)(f1 - (float)h1);
  dlo[4 * t + 2] = (__bf16)(f2 - (float)h2);
  dlo[4 * t + 3] = (__bf16)(f3 - (float)h3);
}

// assemble one 16-bit A/B-matrix WMMA fragment from two aligned 16B chunks
// (ISA 16-bit operand layout: lanes 0-15 hold K kb..kb+7 and kb+16..kb+23)
__device__ __forceinline__ v16bf load_frag(const __bf16* __restrict__ p) {
  const v8bf lo = *(const v8bf*)p;          // K = kb .. kb+7
  const v8bf hi = *(const v8bf*)(p + 16);   // K = kb+16 .. kb+23
  v16bf f;
#pragma unroll
  for (int e = 0; e < 8; ++e) { f[e] = lo[e]; f[8 + e] = hi[e]; }
  return f;
}

// ---------------------------------------------------------------------------
// K2: fused error-compensated WMMA GEMM + soft-DTW per batch pair.
// grid = 1024 (pair p = i*32 + j), block = 256 (8 waves).
// Wave w computes a 32(M) x 64(N) tile: 2x4 WMMA 16x16x32 fragments,
// each accumulated as Ahi*Bhi + Ahi*Blo + Alo*Bhi (f32 accumulate).
// ---------------------------------------------------------------------------
__global__ __launch_bounds__(256) void pair_dtw_kernel(
    const __bf16* __restrict__ imb_hi, const __bf16* __restrict__ imb_lo,
    const __bf16* __restrict__ sb_hi,  const __bf16* __restrict__ sb_lo,
    const int* __restrict__ vlen, const int* __restrict__ clen,
    float* __restrict__ scores) {
  __shared__ float Dlds[MDIM * DSTRIDE];   // 66,560 B cost tile
  __shared__ float dbuf[3][132];           // rotating DP anti-diagonals

  const int p    = blockIdx.x;
  const int bi   = p >> 5;                 // video index
  const int bj   = p & 31;                 // caption index
  const int tid  = threadIdx.x;
  const int lane = tid & 31;
  const int wave = tid >> 5;
  const int mTile = (wave & 3) * 32;       // 4 waves along M
  const int nTile = (wave >> 2) * 64;      // 2 waves along N
  const int l15   = lane & 15;
  const int kb    = (lane < 16) ? 0 : 8;   // ISA 16-bit A-matrix K layout

  v8f acc[2][4];
#pragma unroll
  for (int a = 0; a < 2; ++a)
#pragma unroll
    for (int b = 0; b < 4; ++b)
#pragma unroll
      for (int r = 0; r < 8; ++r) acc[a][b][r] = 0.0f;

  size_t aoff[2], boff[4];
#pragma unroll
  for (int a = 0; a < 2; ++a)
    aoff[a] = (size_t)(bi * MDIM + mTile + 16 * a + l15) * C_DIM + kb;
#pragma unroll
  for (int b = 0; b < 4; ++b)
    boff[b] = (size_t)(bj * NDIM + nTile + 16 * b + l15) * C_DIM + kb;

  for (int kk = 0; kk < C_DIM; kk += 32) {
    v16bf ah[2], al[2], bh[4], bl[4];
#pragma unroll
    for (int a = 0; a < 2; ++a) {
      ah[a] = load_frag(imb_hi + aoff[a] + kk);
      al[a] = load_frag(imb_lo + aoff[a] + kk);
      __builtin_prefetch(imb_hi + aoff[a] + kk + 32, 0, 3);
      __builtin_prefetch(imb_lo + aoff[a] + kk + 32, 0, 3);
    }
#pragma unroll
    for (int b = 0; b < 4; ++b) {
      bh[b] = load_frag(sb_hi + boff[b] + kk);
      bl[b] = load_frag(sb_lo + boff[b] + kk);
      __builtin_prefetch(sb_hi + boff[b] + kk + 32, 0, 3);
      __builtin_prefetch(sb_lo + boff[b] + kk + 32, 0, 3);
    }
#pragma unroll
    for (int a = 0; a < 2; ++a)
#pragma unroll
      for (int b = 0; b < 4; ++b) {
        acc[a][b] = __builtin_amdgcn_wmma_f32_16x16x32_bf16(
            false, ah[a], false, bh[b], (short)0, acc[a][b], false, false);
        acc[a][b] = __builtin_amdgcn_wmma_f32_16x16x32_bf16(
            false, ah[a], false, bl[b], (short)0, acc[a][b], false, false);
        acc[a][b] = __builtin_amdgcn_wmma_f32_16x16x32_bf16(
            false, al[a], false, bh[b], (short)0, acc[a][b], false, false);
      }
  }

  // scatter D = 1 - sim into LDS per the 16x16 f32 C/D VGPR layout
  const int mHalf = (lane >= 16) ? 8 : 0;
#pragma unroll
  for (int a = 0; a < 2; ++a)
#pragma unroll
    for (int b = 0; b < 4; ++b)
#pragma unroll
      for (int r = 0; r < 8; ++r) {
        const int mrow = mTile + 16 * a + mHalf + r;
        const int ncol = nTile + 16 * b + l15;
        Dlds[mrow * DSTRIDE + ncol] = 1.0f - acc[a][b][r];
      }

  // ---- soft-DTW wavefront over the LDS tile -------------------------------
  int la = vlen[bi]; la = la < 1 ? 1 : (la > MDIM ? MDIM : la);
  int lb = clen[bj]; lb = lb < 1 ? 1 : (lb > NDIM ? NDIM : lb);

  if (tid == 0) {                  // diagonals 0 and 1 (padded DP borders)
    dbuf[0][0] = 0.0f;             // R[0][0]
    dbuf[1][0] = BIGF;             // R[0][1]
    dbuf[1][1] = BIGF;             // R[1][0]
  }
  __syncthreads();

  const int ipad = tid + 1;        // padded row handled by this lane (tid<128)
  for (int dd = 2; dd <= MDIM + NDIM; ++dd) {
    float*       cur = dbuf[dd % 3];
    const float* p1  = dbuf[(dd + 2) % 3];   // diagonal dd-1
    const float* p2  = dbuf[(dd + 1) % 3];   // diagonal dd-2
    if (tid < MDIM) {
      const int jp = dd - ipad;
      if (jp >= 1 && jp <= NDIM) {
        const float left = p1[ipad];         // R[i][j-1]
        const float up   = p1[ipad - 1];     // R[i-1][j]
        const float dg   = p2[ipad - 1];     // R[i-1][j-1]
        const float d    = Dlds[(ipad - 1) * DSTRIDE + (jp - 1)];
        const float m    = fminf(left, fminf(up, dg));
        const float ssum = __expf((m - left) * 10.0f) +
                           __expf((m - up)   * 10.0f) +
                           __expf((m - dg)   * 10.0f);
        const float r = d + m - 0.1f * __logf(ssum);
        cur[ipad] = r;
        if (ipad == la && jp == lb) scores[p] = -r * 0.01f;
      }
    } else if (tid == MDIM) {                // maintain BIG borders
      if (dd <= MDIM) { cur[0] = BIGF; cur[dd] = BIGF; }
    }
    __syncthreads();
  }
}

// ---------------------------------------------------------------------------
// K3: contrastive hinge reduction over the 32x32 score matrix -> scalar.
// ---------------------------------------------------------------------------
__global__ __launch_bounds__(256) void loss_kernel(
    const float* __restrict__ scores, float* __restrict__ out) {
  __shared__ float red[256];
  const int t = threadIdx.x;
  float acc = 0.0f;
  for (int idx = t; idx < BDIM * BDIM; idx += 256) {
    const int i = idx >> 5, j = idx & 31;
    if (i != j) {
      const float sij = scores[idx];
      const float di  = scores[i * BDIM + i];
      const float dj  = scores[j * BDIM + j];
      acc += fmaxf(sij - di, 0.0f) + fmaxf(sij - dj, 0.0f);
    }
  }
  red[t] = acc;
  __syncthreads();
#pragma unroll
  for (int k = 128; k > 0; k >>= 1) {
    if (t < k) red[t] += red[t + k];
    __syncthreads();
  }
  if (t == 0) out[0] = red[0];
}

// ---------------------------------------------------------------------------
extern "C" void kernel_launch(void* const* d_in, const int* in_sizes, int n_in,
                              void* d_out, int out_size, void* d_ws, size_t ws_size,
                              hipStream_t stream) {
  (void)in_sizes; (void)n_in; (void)out_size; (void)ws_size;
  const float* im = (const float*)d_in[0];
  const float* s  = (const float*)d_in[1];
  const int* vlen = (const int*)d_in[2];
  const int* clen = (const int*)d_in[3];

  const size_t rowsA = (size_t)BDIM * MDIM;   // 4096
  const size_t plane = rowsA * C_DIM;         // 4M bf16 elements = 8 MB
  __bf16* imb_hi = (__bf16*)d_ws;
  __bf16* imb_lo = imb_hi + plane;
  __bf16* sb_hi  = imb_lo + plane;
  __bf16* sb_lo  = sb_hi + plane;
  float* scores  = (float*)((char*)d_ws + 4u * plane * sizeof(__bf16));

  norm_split_bf16_kernel<<<2 * (int)rowsA, 256, 0, stream>>>(
      im, s, imb_hi, imb_lo, sb_hi, sb_lo);
  pair_dtw_kernel<<<BDIM * BDIM, 256, 0, stream>>>(
      imb_hi, imb_lo, sb_hi, sb_lo, vlen, clen, scores);
  loss_kernel<<<1, 256, 0, stream>>>(scores, (float*)d_out);
}
